// SelfAtt_77223511982529
// MI455X (gfx1250) — matrix operand order repaired and checked by reading, not compiled, not run on hardware
//
#include <hip/hip_runtime.h>

#define BATCH   8
#define NSEQ    2048
#define DMODEL  768
#define QROWS   32
#define CHUNK   64
#define RSTRIDE (DMODEL + 16)   // 784 elems -> 1568B rows (32B aligned)
#define TSTRIDE (CHUNK + 8)     // 72 elems  -> 144B rows (16B aligned)
#define NEGBIG  (-3.0e38f)

typedef __attribute__((ext_vector_type(16))) __bf16 v16bf;
typedef __attribute__((ext_vector_type(8)))  __bf16 v8bf;
typedef __attribute__((ext_vector_type(4)))  __bf16 v4bf;
typedef __attribute__((ext_vector_type(8)))  float  v8f;

static __device__ inline __bf16 f2bf(float f) {
  unsigned u = __builtin_bit_cast(unsigned, f);
  u += 0x7FFFu + ((u >> 16) & 1u);                 // round-to-nearest-even
  unsigned short h = (unsigned short)(u >> 16);
  return __builtin_bit_cast(__bf16, h);
}

// A-fragment (16x32 bf16): lane<16 holds K {k0..k0+7, k0+16..k0+23},
// lane>=16 holds K {k0+8..k0+15, k0+24..k0+31}; row m = lane&15.
static __device__ inline v16bf frag_a(const __bf16* row, int k0, int H) {
  v8bf lo = *(const v8bf*)(row + k0 + H * 8);
  v8bf hi = *(const v8bf*)(row + k0 + 16 + H * 8);
  v16bf r;
#pragma unroll
  for (int e = 0; e < 8; ++e) { r[e] = lo[e]; r[e + 8] = hi[e]; }
  return r;
}

// B-fragment (32x16 bf16): lane<16 = col n, K {k0..k0+7, k0+16..k0+23};
// lane>=16 = col n, K {k0+8..k0+15, k0+24..k0+31}. Storage must be [n][k] contiguous in k.
static __device__ inline v16bf frag_b(const __bf16* row, int k0, int H) {
  v8bf lo = *(const v8bf*)(row + k0 + H * 8);
  v8bf hi = *(const v8bf*)(row + k0 + 16 + H * 8);
  v16bf r;
#pragma unroll
  for (int e = 0; e < 8; ++e) { r[e] = lo[e]; r[e + 8] = hi[e]; }
  return r;
}

__global__ __launch_bounds__(256)
void selfatt_flash_wmma(const float* __restrict__ x,
                        const unsigned char* __restrict__ mask,
                        float* __restrict__ out) {
  __shared__ __bf16 Qs[QROWS][RSTRIDE];   // Q tile, row-major  (A of S=Q*X^T)
  __shared__ __bf16 Xs[CHUNK][RSTRIDE];   // key chunk row-major (B of S)
  __shared__ __bf16 Xt[DMODEL][TSTRIDE];  // key chunk transposed (B of O=P*X)
  __shared__ __bf16 Ps[QROWS][TSTRIDE];   // probs bf16 (A of O)
  __shared__ float  cmax[4][QROWS];
  __shared__ float  csum[4][QROWS];
  __shared__ float  runm[QROWS];
  __shared__ float  runl[QROWS];

  const int tid = threadIdx.x;
  const int w   = tid >> 5;      // wave 0..7
  const int L   = tid & 31;
  const int H   = L >> 4;        // lane half
  const int l   = L & 15;
  const int r   = w & 1;         // row tile (0/1)
  const int kt  = w >> 1;        // key tile in chunk (scores) / D-quarter (PV)
  const int rbase = r * 16;

  const int b  = blockIdx.x >> 6;           // 64 row-blocks per batch
  const int i0 = (blockIdx.x & 63) * QROWS;

  if (tid < QROWS) { runm[tid] = NEGBIG; runl[tid] = 0.0f; }

  // stage Q tile (32x768 f32 -> bf16)
  {
    const float4* gq = (const float4*)(x + ((size_t)b * NSEQ + i0) * DMODEL);
#pragma unroll 6
    for (int q = 0; q < (QROWS * DMODEL / 4) / 256; ++q) {
      int idx = q * 256 + tid;
      float4 t = gq[idx];
      int flat = idx * 4;
      int rr = flat / DMODEL, dd = flat % DMODEL;
      v4bf p = {f2bf(t.x), f2bf(t.y), f2bf(t.z), f2bf(t.w)};
      *(v4bf*)&Qs[rr][dd] = p;
    }
  }

  v8f acc[12];
#pragma unroll
  for (int f = 0; f < 12; ++f)
#pragma unroll
    for (int e = 0; e < 8; ++e) acc[f][e] = 0.0f;

  for (int c = 0; c < NSEQ / CHUNK; ++c) {
    const int j0 = c * CHUNK;
    __syncthreads();  // previous Ps/Xt consumers done before overwrite

    // stage key chunk (64x768): 4x4 register micro-transpose, b64 stores only
    {
      const float4* gk = (const float4*)(x + ((size_t)b * NSEQ + j0) * DMODEL);
#pragma unroll 4
      for (int g = 0; g < (CHUNK / 4) * (DMODEL / 4) / 256; ++g) {  // 12
        int qq = g * 256 + tid;         // 4x4 block index
        int jq = qq / (DMODEL / 4);     // row quad 0..15
        int dq = qq % (DMODEL / 4);     // col quad 0..191
        float4 t0 = gk[(jq * 4 + 0) * (DMODEL / 4) + dq];
        float4 t1 = gk[(jq * 4 + 1) * (DMODEL / 4) + dq];
        float4 t2 = gk[(jq * 4 + 2) * (DMODEL / 4) + dq];
        float4 t3 = gk[(jq * 4 + 3) * (DMODEL / 4) + dq];
        // row-major copy (Xs): one b64 store per source row
        v4bf r0 = {f2bf(t0.x), f2bf(t0.y), f2bf(t0.z), f2bf(t0.w)};
        v4bf r1 = {f2bf(t1.x), f2bf(t1.y), f2bf(t1.z), f2bf(t1.w)};
        v4bf r2 = {f2bf(t2.x), f2bf(t2.y), f2bf(t2.z), f2bf(t2.w)};
        v4bf r3 = {f2bf(t3.x), f2bf(t3.y), f2bf(t3.z), f2bf(t3.w)};
        *(v4bf*)&Xs[jq * 4 + 0][dq * 4] = r0;
        *(v4bf*)&Xs[jq * 4 + 1][dq * 4] = r1;
        *(v4bf*)&Xs[jq * 4 + 2][dq * 4] = r2;
        *(v4bf*)&Xs[jq * 4 + 3][dq * 4] = r3;
        // transposed copy (Xt): one b64 store per source column
        v4bf c0 = {r0[0], r1[0], r2[0], r3[0]};
        v4bf c1 = {r0[1], r1[1], r2[1], r3[1]};
        v4bf c2 = {r0[2], r1[2], r2[2], r3[2]};
        v4bf c3 = {r0[3], r1[3], r2[3], r3[3]};
        *(v4bf*)&Xt[dq * 4 + 0][jq * 4] = c0;
        *(v4bf*)&Xt[dq * 4 + 1][jq * 4] = c1;
        *(v4bf*)&Xt[dq * 4 + 2][jq * 4] = c2;
        *(v4bf*)&Xt[dq * 4 + 3][jq * 4] = c3;
      }
    }

    // prefetch next key chunk into cache (global_prefetch_b8); L2-resident data
    if (c + 1 < NSEQ / CHUNK) {
      const char* nx =
          (const char*)(x + ((size_t)b * NSEQ + j0 + CHUNK) * DMODEL);
#pragma unroll
      for (int q = 0; q < 6; ++q)
        __builtin_prefetch(nx + (q * 256 + tid) * 128, 0, 1);
    }
    __syncthreads();

    // ---- scores: S[16x16] for (row tile r, key tile kt), K-reduce over D ----
    // two independent WMMA accumulation chains for ILP
    v8f s, s1;
#pragma unroll
    for (int e = 0; e < 8; ++e) { s[e] = 0.0f; s1[e] = 0.0f; }
    const __bf16* arow = &Qs[rbase + l][0];
    const __bf16* brow = &Xs[kt * 16 + l][0];
#pragma unroll
    for (int d0 = 0; d0 < DMODEL; d0 += 64) {
      v16bf a0  = frag_a(arow, d0, H);
      v16bf bb0 = frag_b(brow, d0, H);
      s = __builtin_amdgcn_wmma_f32_16x16x32_bf16(false, a0, false, bb0,
                                                  (short)0, s, false, false);
      v16bf a1  = frag_a(arow, d0 + 32, H);
      v16bf bb1 = frag_b(brow, d0 + 32, H);
      s1 = __builtin_amdgcn_wmma_f32_16x16x32_bf16(false, a1, false, bb1,
                                                   (short)0, s1, false, false);
    }
#pragma unroll
    for (int e = 0; e < 8; ++e) s[e] += s1[e];

    // masking: diagonal + padded keys (col j fixed per lane)
    const int jG = j0 + kt * 16 + l;
    const bool pad = mask[(size_t)b * NSEQ + jG] != 0;
#pragma unroll
    for (int v = 0; v < 8; ++v) {
      int iG = i0 + rbase + v + 8 * H;
      if (pad || (jG == iG)) s[v] = NEGBIG;
    }

    // per-row partial max across 16 lanes (xor masks stay inside half-groups)
    float m8[8];
#pragma unroll
    for (int v = 0; v < 8; ++v) {
      float mv = s[v];
      mv = fmaxf(mv, __shfl_xor(mv, 1));
      mv = fmaxf(mv, __shfl_xor(mv, 2));
      mv = fmaxf(mv, __shfl_xor(mv, 4));
      mv = fmaxf(mv, __shfl_xor(mv, 8));
      m8[v] = mv;
    }
    if (l == 0) {
#pragma unroll
      for (int v = 0; v < 8; ++v) cmax[kt][rbase + v + 8 * H] = m8[v];
    }
    __syncthreads();

    // online softmax: combine chunk max, exponentiate, partial sums, P -> LDS
    float mnew8[8], scale8[8], sum8[8];
#pragma unroll
    for (int v = 0; v < 8; ++v) {
      int row = rbase + v + 8 * H;
      float mc = fmaxf(fmaxf(cmax[0][row], cmax[1][row]),
                       fmaxf(cmax[2][row], cmax[3][row]));
      float mold = runm[row];
      float mn = fmaxf(mold, mc);
      mnew8[v]  = mn;
      scale8[v] = __expf(mold - mn);
      float pv  = __expf(s[v] - mn);
      float sv = pv;
      sv += __shfl_xor(sv, 1);
      sv += __shfl_xor(sv, 2);
      sv += __shfl_xor(sv, 4);
      sv += __shfl_xor(sv, 8);
      sum8[v] = sv;
      Ps[row][kt * 16 + l] = f2bf(pv);
    }
    if (l == 0) {
#pragma unroll
      for (int v = 0; v < 8; ++v) csum[kt][rbase + v + 8 * H] = sum8[v];
    }
    __syncthreads();

    if (kt == 0 && l == 0) {
#pragma unroll
      for (int v = 0; v < 8; ++v) {
        int row = rbase + v + 8 * H;
        runl[row] = runl[row] * scale8[v] +
                    csum[0][row] + csum[1][row] + csum[2][row] + csum[3][row];
        runm[row] = mnew8[v];
      }
    }

    // rescale running accumulators (rows match scale8's row pattern)
#pragma unroll
    for (int f = 0; f < 12; ++f)
#pragma unroll
      for (int v = 0; v < 8; ++v) acc[f][v] *= scale8[v];

    // ---- O += P * X : wave owns (row tile r) x (D-quarter kt), K = 64 keys ----
    const __bf16* prow = &Ps[rbase + l][0];
#pragma unroll
    for (int kk = 0; kk < CHUNK; kk += 32) {
      v16bf a = frag_a(prow, kk, H);
#pragma unroll
      for (int f = 0; f < 12; ++f) {
        const __bf16* vrow = &Xt[kt * 192 + f * 16 + l][0];
        v16bf bb = frag_b(vrow, kk, H);
        acc[f] = __builtin_amdgcn_wmma_f32_16x16x32_bf16(false, a, false, bb,
                                                         (short)0, acc[f],
                                                         false, false);
      }
    }
  }

  __syncthreads();
  float inv8[8];
#pragma unroll
  for (int v = 0; v < 8; ++v) inv8[v] = 1.0f / runl[rbase + v + 8 * H];
#pragma unroll
  for (int f = 0; f < 12; ++f) {
    int dcol = kt * 192 + f * 16 + l;
#pragma unroll
    for (int v = 0; v < 8; ++v) {
      int row = rbase + v + 8 * H;
      out[((size_t)b * NSEQ + i0 + row) * DMODEL + dcol] = acc[f][v] * inv8[v];
    }
  }
}

extern "C" void kernel_launch(void* const* d_in, const int* in_sizes, int n_in,
                              void* d_out, int out_size, void* d_ws, size_t ws_size,
                              hipStream_t stream) {
  (void)in_sizes; (void)n_in; (void)out_size; (void)d_ws; (void)ws_size;
  const float* x = (const float*)d_in[0];
  const unsigned char* mask = (const unsigned char*)d_in[1];  // jax bool -> 1 byte
  float* out = (float*)d_out;
  dim3 grid(BATCH * (NSEQ / QROWS));   // 8 * 64 = 512 blocks
  dim3 block(256);                     // 8 waves of 32
  selfatt_flash_wmma<<<grid, block, 0, stream>>>(x, mask, out);
}
